// MultiHeadAttention_8658654069207
// MI455X (gfx1250) — matrix operand order repaired and checked by reading, not compiled, not run on hardware
//
#include <hip/hip_runtime.h>
#include <stdint.h>

// ---------------------------------------------------------------------------
// CDNA5 (gfx1250) bf16 WMMA multi-head attention.
// v_wmma_f32_16x16x32_bf16 + TDM tensor_load_to_lds + async global->LDS,
// double-buffered tiles.
// ---------------------------------------------------------------------------

typedef __attribute__((ext_vector_type(16))) __bf16 v16bf;
typedef __attribute__((ext_vector_type(8)))  float  v8f;
typedef unsigned int u32x4 __attribute__((ext_vector_type(4)));
typedef int          i32x8 __attribute__((ext_vector_type(8)));
typedef int          i32x4 __attribute__((ext_vector_type(4)));

#define WMMA_BF16(a, b, c) \
  __builtin_amdgcn_wmma_f32_16x16x32_bf16(false, (a), false, (b), (short)0, (c), false, false)

__device__ __forceinline__ uint16_t f2bf(float f) {
  union { float f; uint32_t u; } v; v.f = f;
  uint32_t r = v.u + 0x7FFFu + ((v.u >> 16) & 1u);   // round-to-nearest-even
  return (uint16_t)(r >> 16);
}

// --- gfx1250 data movers ----------------------------------------------------
// Async global->LDS 16-byte copy (ASYNCcnt). LDS byte offset = low 32 bits of
// the generic pointer to a __shared__ object (flat aperture truncation).
__device__ __forceinline__ void async_ld_b128(uint16_t* lds_dst, const uint16_t* gsrc) {
  uint32_t l = (uint32_t)(uintptr_t)lds_dst;
  asm volatile("global_load_async_to_lds_b128 %0, %1, off" :: "v"(l), "v"(gsrc) : "memory");
}
__device__ __forceinline__ void wait_async0() {
  asm volatile("s_wait_asynccnt 0" ::: "memory");
}
__device__ __forceinline__ void wait_ds0() {
  asm volatile("s_wait_dscnt 0" ::: "memory");
}
__device__ __forceinline__ void wait_tensor0() {
  __builtin_amdgcn_s_wait_tensorcnt(0);
}

// TDM: one instruction moves a 2-D strided tile (bf16) global->LDS.
// D# per CDNA5 ISA §8: group0 = {count|flags, lds_addr, global_addr, type=2},
// group1 = {data_size=1(2B), tensor dims, tile dims, dim0 stride}.
// This toolchain exposes the 6-arg builtin: (g0, g1, g2, g3, g_extra, cpol).
__device__ __forceinline__ void tdm_load_2d(uint16_t* lds_dst, const void* gaddr,
                                            int tile_d0, int tile_d1, int stride0) {
  uint64_t ga = (uint64_t)(uintptr_t)gaddr;
  u32x4 g0;
  g0.x = 1u;                                       // count=1 (valid), user mode
  g0.y = (uint32_t)(uintptr_t)lds_dst;             // lds_addr (bytes)
  g0.z = (uint32_t)ga;                             // global_addr[31:0]
  g0.w = (uint32_t)(ga >> 32) | (2u << 30);        // global_addr[56:32] | type=2
  i32x8 g1;
  g1[0] = (1 << 16);                               // data_size=1 -> 2 bytes
  g1[1] = (tile_d0 & 0xFFFF) << 16;                // tensor_dim0[15:0]   (bits 63:48)
  g1[2] = ((unsigned)tile_d0 >> 16) |              // tensor_dim0[31:16]  (bits 79:64)
          ((tile_d1 & 0xFFFF) << 16);              // tensor_dim1[15:0]   (bits 95:80)
  g1[3] = ((unsigned)tile_d1 >> 16) |              // tensor_dim1[31:16]  (bits 111:96)
          ((tile_d0 & 0xFFFF) << 16);              // tile_dim0           (bits 127:112)
  g1[4] = (tile_d1 & 0xFFFF);                      // tile_dim1 (tile_dim2=0)
  g1[5] = stride0;                                 // tensor_dim0_stride[31:0]
  g1[6] = 0;                                       // stride0[47:32] | dim1_stride lo
  g1[7] = 0;
  i32x4 z4 = {0, 0, 0, 0};
  i32x8 z8 = {0, 0, 0, 0, 0, 0, 0, 0};
  __builtin_amdgcn_tensor_load_to_lds(g0, g1, z4, z4, z8, 0);
}

// --- WMMA fragment loaders --------------------------------------------------
union FragB16 { v16bf v; uint32_t u[8]; };

// A-fragment (16x32 bf16): row = lane&15; ISA 16-bit A layout:
//   j=0..7 -> K = hi*8 + j ; j=8..15 -> K = 16 + hi*8 + (j-8)
__device__ __forceinline__ v16bf load_a_frag(const uint16_t* base, int stride) {
  const int lane = threadIdx.x & 31;
  const int lo = lane & 15, hi = lane >> 4;
  const uint32_t* p = (const uint32_t*)(base + lo * stride);
  FragB16 f;
  const int k0 = hi * 4;
#pragma unroll
  for (int i = 0; i < 4; ++i) {
    f.u[i]     = p[k0 + i];
    f.u[4 + i] = p[8 + k0 + i];
  }
  return f.v;
}

// B-fragment (32x16 bf16) from n-major LDS [n][k]: col = lane&15,
//   j=0..7 -> K = hi*8 + j ; j=8..15 -> K = 16 + hi*8 + (j-8)
__device__ __forceinline__ v16bf load_b_frag(const uint16_t* base, int stride) {
  const int lane = threadIdx.x & 31;
  const int lo = lane & 15, hi = lane >> 4;
  const uint32_t* p = (const uint32_t*)(base + lo * stride);
  FragB16 f;
  const int k0 = hi * 4;
#pragma unroll
  for (int i = 0; i < 4; ++i) {
    f.u[i]     = p[k0 + i];
    f.u[4 + i] = p[8 + k0 + i];
  }
  return f.v;
}

// ---------------------------------------------------------------------------
// Tiled GEMM: C[M,N] = A[M,Kd]*B[Kd,N], bf16 in, fp32 accum.
// 256 threads (8 waves). Block tile 128x128, wave tile 32x64 (8 WMMA/k-step).
// A tiles: TDM tensor_load_to_lds (one instruction per tile, wave 0), double
// buffered. B tiles: register-staged transpose to n-major LDS, pipelined.
// ---------------------------------------------------------------------------
template <bool OUT_BF16>
__global__ void __launch_bounds__(256)
gemm_bf16_kernel(const uint16_t* __restrict__ A, const uint16_t* __restrict__ B,
                 void* __restrict__ Cv, int M, int N, int Kd) {
  __shared__ uint16_t sA[2][128 * 32];    // [m][k]
  __shared__ uint16_t sB[2][128 * 32];    // [n][k] (transposed on fill)
  const int tid  = threadIdx.x;
  const int wave = tid >> 5;
  const int lane = tid & 31, lo = lane & 15, hi = lane >> 4;
  const int nb = blockIdx.x * 128;
  const int mb = blockIdx.y * 128;
  const int m0 = (wave & 3) * 32;
  const int n0 = (wave >> 2) * 64;

  const v8f vzero = {0.f,0.f,0.f,0.f,0.f,0.f,0.f,0.f};
  v8f acc0[4] = {vzero, vzero, vzero, vzero};
  v8f acc1[4] = {vzero, vzero, vzero, vzero};
  uint32_t breg[8];

  auto fill_A = [&](int k0, uint16_t* buf) {       // 128x32 tile via TDM
    if (wave == 0)
      tdm_load_2d(buf, A + (size_t)mb * Kd + k0, 32, 128, Kd);
  };
  auto load_B_regs = [&](int k0) {                 // 32x128 tile, coalesced dwords
#pragma unroll
    for (int i = 0; i < 8; ++i) {
      int e = (tid * 8 + i) * 2;
      int kr = e >> 7, nc = e & 127;
      breg[i] = *(const uint32_t*)(B + (size_t)(k0 + kr) * N + nb + nc);
    }
  };
  auto store_B = [&](uint16_t* buf) {              // transpose to [n][k]
#pragma unroll
    for (int i = 0; i < 8; ++i) {
      int e = (tid * 8 + i) * 2;
      int kr = e >> 7, nc = e & 127;
      buf[(size_t)nc * 32 + kr]       = (uint16_t)(breg[i] & 0xFFFFu);
      buf[(size_t)(nc + 1) * 32 + kr] = (uint16_t)(breg[i] >> 16);
    }
  };

  fill_A(0, sA[0]);
  load_B_regs(0);
  store_B(sB[0]);

  const int nsteps = Kd >> 5;
  for (int i = 0; i < nsteps; ++i) {
    const int cur = i & 1, nxt = cur ^ 1;
    if (wave == 0) wait_tensor0();                 // tile i DMA complete
    __syncthreads();                               // tile i visible to all waves
    if (i + 1 < nsteps) {                          // start tile i+1 fills
      fill_A((i + 1) * 32, sA[nxt]);
      load_B_regs((i + 1) * 32);
    }
    v16bf a0 = load_a_frag(sA[cur] + m0 * 32, 32);
    v16bf a1 = load_a_frag(sA[cur] + (m0 + 16) * 32, 32);
#pragma unroll
    for (int t = 0; t < 4; ++t) {
      v16bf b = load_b_frag(sB[cur] + (n0 + t * 16) * 32, 32);
      acc0[t] = WMMA_BF16(a0, b, acc0[t]);
      acc1[t] = WMMA_BF16(a1, b, acc1[t]);
    }
    if (i + 1 < nsteps) store_B(sB[nxt]);
  }

#pragma unroll
  for (int t = 0; t < 4; ++t) {
#pragma unroll
    for (int r = 0; r < 8; ++r) {
      int col = nb + n0 + t * 16 + lo;
      int row0 = mb + m0 + hi * 8 + r;             // C/D: M = elem + hi*8
      int row1 = row0 + 16;
      if constexpr (OUT_BF16) {
        ((uint16_t*)Cv)[(size_t)row0 * N + col] = f2bf(acc0[t][r]);
        ((uint16_t*)Cv)[(size_t)row1 * N + col] = f2bf(acc1[t][r]);
      } else {
        ((float*)Cv)[(size_t)row0 * N + col] = acc0[t][r];
        ((float*)Cv)[(size_t)row1 * N + col] = acc1[t][r];
      }
    }
  }
}

// ---------------------------------------------------------------------------
// Flash attention, one (b, h, 64-query tile) per 128-thread block. d_head=64.
// Layout: [(b*S + s)*(H*64) + h*64 + d], bf16. Q: async->LDS; K: TDM, double
// buffered; V: register-staged transpose, double buffered.
// ---------------------------------------------------------------------------
__global__ void __launch_bounds__(128)
attn_kernel(const uint16_t* __restrict__ Q, const uint16_t* __restrict__ Km,
            const uint16_t* __restrict__ V, uint16_t* __restrict__ ctx,
            int S, int H) {
  __shared__ uint16_t sQ[64 * 64];        // [q][d]
  __shared__ uint16_t sP[64 * 64];        // [q][key] probs
  __shared__ uint16_t sK[2][64 * 64];     // [key][d] (n-major B for Q*K^T)
  __shared__ uint16_t sV[2][64 * 64];     // [d][key] (n-major B for P*V)

  const int tid  = threadIdx.x;
  const int wave = tid >> 5;
  const int lane = tid & 31, lo = lane & 15, hi = lane >> 4;
  const int qb = blockIdx.x * 64;
  const int h  = blockIdx.y;
  const int bb = blockIdx.z;
  const int ld = H * 64;
  const size_t headOff = (size_t)bb * S * ld + (size_t)h * 64;
  const int m0 = wave * 16;

  // Q tile: per-lane async global->LDS (512 x 16B chunks, 128 threads x 4)
#pragma unroll
  for (int t = 0; t < 4; ++t) {
    int c = tid + t * 128;
    int r = c >> 3, cc = (c & 7) << 3;
    async_ld_b128(sQ + r * 64 + cc, Q + headOff + (size_t)(qb + r) * ld + cc);
  }
  auto fill_K = [&](int kb, uint16_t* buf) {       // 64x64 tile via TDM
    if (wave == 0)
      tdm_load_2d(buf, Km + headOff + (size_t)kb * ld, 64, 64, ld);
  };
  uint32_t vreg[16];
  auto load_V_regs = [&](int kb) {
#pragma unroll
    for (int i = 0; i < 16; ++i) {
      int e = (tid * 16 + i) * 2;
      int r = e >> 6, c = e & 63;
      vreg[i] = *(const uint32_t*)(V + headOff + (size_t)(kb + r) * ld + c);
    }
  };
  auto store_V = [&](uint16_t* buf) {              // transpose -> [d][key]
#pragma unroll
    for (int i = 0; i < 16; ++i) {
      int e = (tid * 16 + i) * 2;
      int r = e >> 6, c = e & 63;
      buf[(size_t)c * 64 + r]       = (uint16_t)(vreg[i] & 0xFFFFu);
      buf[(size_t)(c + 1) * 64 + r] = (uint16_t)(vreg[i] >> 16);
    }
  };

  fill_K(0, sK[0]);
  load_V_regs(0);
  store_V(sV[0]);

  const v8f vzero = {0.f,0.f,0.f,0.f,0.f,0.f,0.f,0.f};
  v8f o[4] = {vzero, vzero, vzero, vzero};
  float mrun[8], lrun[8];
#pragma unroll
  for (int r = 0; r < 8; ++r) { mrun[r] = -3.0e38f; lrun[r] = 0.f; }

  const int nkb = S >> 6;
  for (int i = 0; i < nkb; ++i) {
    const int cur = i & 1, nxt = cur ^ 1;
    wait_async0();                                 // Q fill (iter 0)
    if (wave == 0) wait_tensor0();                 // K tile i DMA complete
    __syncthreads();
    if (i + 1 < nkb) {
      fill_K((i + 1) * 64, sK[nxt]);
      load_V_regs((i + 1) * 64);
    }

    // scores = Q(16x64) * K^T(64x64) per wave
    v8f sc[4] = {vzero, vzero, vzero, vzero};
#pragma unroll
    for (int kc = 0; kc < 64; kc += 32) {
      v16bf a = load_a_frag(sQ + m0 * 64 + kc, 64);
#pragma unroll
      for (int t = 0; t < 4; ++t) {
        v16bf b = load_b_frag(sK[cur] + (t * 16) * 64 + kc, 64);
        sc[t] = WMMA_BF16(a, b, sc[t]);
      }
    }

    // online softmax; elem r is row (m0 + hi*8 + r), 16-lane halves share rows
#pragma unroll
    for (int r = 0; r < 8; ++r) {
      float s0 = sc[0][r] * 0.125f, s1 = sc[1][r] * 0.125f;
      float s2 = sc[2][r] * 0.125f, s3 = sc[3][r] * 0.125f;
      float rm = fmaxf(fmaxf(s0, s1), fmaxf(s2, s3));
#pragma unroll
      for (int d = 8; d >= 1; d >>= 1) rm = fmaxf(rm, __shfl_xor(rm, d));
      float mnew = fmaxf(mrun[r], rm);
      float p0 = __expf(s0 - mnew), p1 = __expf(s1 - mnew);
      float p2 = __expf(s2 - mnew), p3 = __expf(s3 - mnew);
      float rs = p0 + p1 + p2 + p3;
#pragma unroll
      for (int d = 8; d >= 1; d >>= 1) rs += __shfl_xor(rs, d);
      float corr = __expf(mrun[r] - mnew);
      lrun[r] = lrun[r] * corr + rs;
      mrun[r] = mnew;
#pragma unroll
      for (int t = 0; t < 4; ++t) o[t][r] *= corr;
      int prow = m0 + hi * 8 + r;
      sP[(size_t)prow * 64 +  0 + lo] = f2bf(p0);
      sP[(size_t)prow * 64 + 16 + lo] = f2bf(p1);
      sP[(size_t)prow * 64 + 32 + lo] = f2bf(p2);
      sP[(size_t)prow * 64 + 48 + lo] = f2bf(p3);
    }
    wait_ds0();                                    // P stores before A-frag reload

    // O += P(16x64) * V(64x64)
#pragma unroll
    for (int kc = 0; kc < 64; kc += 32) {
      v16bf a = load_a_frag(sP + m0 * 64 + kc, 64);
#pragma unroll
      for (int t = 0; t < 4; ++t) {
        v16bf b = load_b_frag(sV[cur] + (t * 16) * 64 + kc, 64);
        o[t] = WMMA_BF16(a, b, o[t]);
      }
    }
    if (i + 1 < nkb) store_V(sV[nxt]);
  }

#pragma unroll
  for (int r = 0; r < 8; ++r) {
    float inv = 1.0f / lrun[r];
    int srow = qb + m0 + hi * 8 + r;
#pragma unroll
    for (int t = 0; t < 4; ++t)
      ctx[headOff + (size_t)srow * ld + t * 16 + lo] = f2bf(o[t][r] * inv);
  }
}

// ---------------------------------------------------------------------------
// Conversion / packing kernels
// ---------------------------------------------------------------------------
__global__ void cvt_kernel(const float* __restrict__ in, uint16_t* __restrict__ out, int n) {
  int i = blockIdx.x * blockDim.x + threadIdx.x;
  if (i < n) out[i] = f2bf(in[i]);
}

// W[H][D][Kh] f32 -> out[D][H*Kh] bf16 (fuse per-head projections into one GEMM)
__global__ void packw_kernel(const float* __restrict__ W, uint16_t* __restrict__ out,
                             int H, int D, int Kh) {
  int i = blockIdx.x * blockDim.x + threadIdx.x;
  int n = H * D * Kh;
  if (i >= n) return;
  int h   = i / (D * Kh);
  int rem = i - h * (D * Kh);
  int d   = rem / Kh;
  int k   = rem - d * Kh;
  out[(size_t)d * (H * Kh) + h * Kh + k] = f2bf(W[i]);
}

// ---------------------------------------------------------------------------
extern "C" void kernel_launch(void* const* d_in, const int* in_sizes, int n_in,
                              void* d_out, int out_size, void* d_ws, size_t ws_size,
                              hipStream_t stream) {
  const float* x  = (const float*)d_in[0];
  const float* Wq = (const float*)d_in[1];
  const float* Wk = (const float*)d_in[2];
  const float* Wv = (const float*)d_in[3];
  const float* Wo = (const float*)d_in[4];

  constexpr int B = 2, S = 2048, D = 1024, H = 16, Kh = 64;
  constexpr int M = B * S;     // 4096
  constexpr int N = H * Kh;    // 1024

  uint8_t* ws = (uint8_t*)d_ws;
  size_t off = 0;
  auto alloc = [&](size_t bytes) -> void* {
    void* p = ws + off;
    off += (bytes + 255) & ~(size_t)255;
    return p;
  };
  uint16_t* xb  = (uint16_t*)alloc((size_t)M * D * 2);
  uint16_t* wqb = (uint16_t*)alloc((size_t)D * N * 2);
  uint16_t* wkb = (uint16_t*)alloc((size_t)D * N * 2);
  uint16_t* wvb = (uint16_t*)alloc((size_t)D * N * 2);
  uint16_t* wob = (uint16_t*)alloc((size_t)N * D * 2);
  uint16_t* Qb  = (uint16_t*)alloc((size_t)M * N * 2);
  uint16_t* Kb  = (uint16_t*)alloc((size_t)M * N * 2);
  uint16_t* Vb  = (uint16_t*)alloc((size_t)M * N * 2);
  uint16_t* Cb  = (uint16_t*)alloc((size_t)M * N * 2);

  cvt_kernel<<<(M * D + 255) / 256, 256, 0, stream>>>(x, xb, M * D);
  packw_kernel<<<(H * D * Kh + 255) / 256, 256, 0, stream>>>(Wq, wqb, H, D, Kh);
  packw_kernel<<<(H * D * Kh + 255) / 256, 256, 0, stream>>>(Wk, wkb, H, D, Kh);
  packw_kernel<<<(H * D * Kh + 255) / 256, 256, 0, stream>>>(Wv, wvb, H, D, Kh);
  cvt_kernel<<<(H * Kh * D + 255) / 256, 256, 0, stream>>>(Wo, wob, H * Kh * D);

  dim3 gg(N / 128, M / 128);
  gemm_bf16_kernel<true><<<gg, 256, 0, stream>>>(xb, wqb, Qb, M, N, D);
  gemm_bf16_kernel<true><<<gg, 256, 0, stream>>>(xb, wkb, Kb, M, N, D);
  gemm_bf16_kernel<true><<<gg, 256, 0, stream>>>(xb, wvb, Vb, M, N, D);

  dim3 ga(S / 64, H, B);
  attn_kernel<<<ga, 128, 0, stream>>>(Qb, Kb, Vb, Cb, S, H);

  dim3 go(D / 128, M / 128);
  gemm_bf16_kernel<false><<<go, 256, 0, stream>>>(Cb, wob, d_out, M, D, N);
}
